// MHAttention_25048249270782
// MI455X (gfx1250) — matrix operand order repaired and checked by reading, not compiled, not run on hardware
//
#include <hip/hip_runtime.h>
#include <stdint.h>

#define DIMW   1024
#define HEADS  16
#define DEPTH  64
#define SEQ    2048
#define BATCH  4

typedef _Float16 h16;
typedef __attribute__((ext_vector_type(16))) _Float16 v16h;
typedef __attribute__((ext_vector_type(8)))  float    v8f;
typedef __attribute__((__vector_size__(16))) int      vi4;

union Frag16 { v16h v; uint4 q[2]; uint32_t u[8]; };
union Acc8   { v8f  v; float f[8]; };
union H2     { uint32_t u; h16 h[2]; };
union V16    { uint4 q[2]; h16 hh[16]; };

// gfx1250 async global->LDS path (ASYNCcnt), guarded so either toolchain builds.
#if defined(__AMDGCN__) && __has_builtin(__builtin_amdgcn_global_load_async_to_lds_b128) && __has_builtin(__builtin_amdgcn_s_wait_asynccnt)
#define HAVE_ASYNC_LDS 1
#define GLB_AS __attribute__((address_space(1)))
#define LDS_AS __attribute__((address_space(3)))
#endif

__device__ __forceinline__ void async_copy_b128x2(const void* gsrc, void* ldst) {
#if HAVE_ASYNC_LDS
  GLB_AS vi4* g = (GLB_AS vi4*)(void*)gsrc;
  LDS_AS vi4* l = (LDS_AS vi4*)ldst;
  __builtin_amdgcn_global_load_async_to_lds_b128(g, l, 0, 0);
  __builtin_amdgcn_global_load_async_to_lds_b128(g, l, 16, 0);
#else
  ((uint4*)ldst)[0] = ((const uint4*)gsrc)[0];
  ((uint4*)ldst)[1] = ((const uint4*)gsrc)[1];
#endif
}

__device__ __forceinline__ void async_wait0() {
#if HAVE_ASYNC_LDS
  __builtin_amdgcn_s_wait_asynccnt(0);
#endif
}

__device__ __forceinline__ v8f wmma_f16(v16h a, v16h b, v8f c) {
  return __builtin_amdgcn_wmma_f32_16x16x32_f16(false, a, false, b, (short)0, c, false, false);
}

__device__ __forceinline__ uint32_t pack2(float a, float b) {
  H2 t; t.h[0] = (h16)a; t.h[1] = (h16)b; return t.u;
}

// A fragment (16x32 f16): per lane, halves {0..7}+hi*8 and {16..23}+hi*8 from
// the lane's M row -> two 16B chunks (ds_load_b128 / global_load_b128).
__device__ __forceinline__ v16h load_afrag(const h16* row, int hi) {
  Frag16 f;
  const uint4* p = (const uint4*)(row + hi * 8);
  f.q[0] = p[0];   // K 0..7   (lanes16-31: 8..15)
  f.q[1] = p[2];   // K 16..23 (lanes16-31: 24..31)
  return f.v;
}

// B fragment (32x16 f16) from [n][k]-major LDS: per lane, halves {0..7}+hi*8
// and {16..23}+hi*8 at column row n -> two 16B chunks (same k order as A).
__device__ __forceinline__ v16h load_bfrag(const h16* row, int hi) {
  Frag16 f;
  const uint4* p = (const uint4*)(row + hi * 8);
  f.q[0] = p[0];
  f.q[1] = p[2];
  return f.v;
}

// ---------------------------------------------------------------------------
// WMMA GEMM: C[r,c] = sum_k X[r,k] * W[k,c], K = N = 1024.
// Block tile 64 rows x 128 cols, k-step 64, 256 threads = 8 waves.
// ---------------------------------------------------------------------------
template <bool IN_F32, bool OUT_HEADS>
__global__ __launch_bounds__(256) void gemm_wmma_kernel(
    const void* __restrict__ xin, const float* __restrict__ w,
    void* __restrict__ outp) {
  constexpr int XS = 72;
  __shared__ h16 lds_x[64 * XS];    // X tile  [row][k]
  __shared__ h16 lds_w[128 * XS];   // W tile transposed [col][k]

  const int c0   = blockIdx.x * 128;
  const int r0   = blockIdx.y * 64;
  const int tid  = threadIdx.x;
  const int lane = tid & 31;
  const int wv   = tid >> 5;
  const int wr   = wv & 3;
  const int cq   = wv >> 2;
  const int nn   = lane & 15;
  const int hi   = lane >> 4;

  Acc8 acc[4];
#pragma unroll
  for (int t = 0; t < 4; ++t) acc[t].v = (v8f){};

  for (int kb = 0; kb < DIMW; kb += 64) {
    // ---- stage X tile (64 rows x 64 halves) ----
    {
      const int row = tid >> 2, seg = tid & 3;
      if (IN_F32) {
        const float* src = (const float*)xin + (size_t)(r0 + row) * DIMW + kb + seg * 16;
        Frag16 pk;
#pragma unroll
        for (int i = 0; i < 4; ++i) {
          float4 f = ((const float4*)src)[i];
          pk.u[2 * i + 0] = pack2(f.x, f.y);
          pk.u[2 * i + 1] = pack2(f.z, f.w);
        }
        uint4* dst = (uint4*)&lds_x[row * XS + seg * 16];
        dst[0] = pk.q[0];
        dst[1] = pk.q[1];
        if (kb + 64 < DIMW) __builtin_prefetch(src + 64, 0, 0);
      } else {
        const h16* src = (const h16*)xin + (size_t)(r0 + row) * DIMW + kb + seg * 16;
        async_copy_b128x2(src, &lds_x[row * XS + seg * 16]);
        if (kb + 64 < DIMW) __builtin_prefetch(src + 64, 0, 0);
      }
    }
    // ---- stage W tile transposed: 2 consecutive k rows x 16 cols/thread ----
    {
      const int kr = (tid >> 3) * 2, cs = tid & 7;
      const float* s0 = w + (size_t)(kb + kr) * DIMW + c0 + cs * 16;
      const float* s1 = s0 + DIMW;
      float r0v[16], r1v[16];
#pragma unroll
      for (int i = 0; i < 16; i += 4) {
        float4 f0 = *(const float4*)(s0 + i);
        float4 f1 = *(const float4*)(s1 + i);
        r0v[i] = f0.x; r0v[i + 1] = f0.y; r0v[i + 2] = f0.z; r0v[i + 3] = f0.w;
        r1v[i] = f1.x; r1v[i + 1] = f1.y; r1v[i + 2] = f1.z; r1v[i + 3] = f1.w;
      }
      uint32_t* lw32 = (uint32_t*)lds_w;
#pragma unroll
      for (int i = 0; i < 16; ++i) {
        const int col = cs * 16 + i;
        lw32[(col * XS + kr) >> 1] = pack2(r0v[i], r1v[i]);
      }
      if (kb + 64 < DIMW) __builtin_prefetch(s0 + 64 * DIMW, 0, 0);
    }
    if (!IN_F32) async_wait0();
    __syncthreads();

    // ---- compute: 8 WMMAs per wave per k-step ----
    const h16* xrow = &lds_x[(wr * 16 + nn) * XS];
    v16h a0 = load_afrag(xrow, hi);
    v16h a1 = load_afrag(xrow + 32, hi);
#pragma unroll
    for (int ct = 0; ct < 4; ++ct) {
      const h16* wrow = &lds_w[(cq * 64 + ct * 16 + nn) * XS];
      v16h b0 = load_bfrag(wrow, hi);
      v16h b1 = load_bfrag(wrow + 32, hi);
      acc[ct].v = wmma_f16(a0, b0, acc[ct].v);
      acc[ct].v = wmma_f16(a1, b1, acc[ct].v);
    }
    __syncthreads();
  }

  // ---- epilogue ----
#pragma unroll
  for (int ct = 0; ct < 4; ++ct) {
    const int c = c0 + cq * 64 + ct * 16 + nn;
#pragma unroll
    for (int e = 0; e < 8; ++e) {
      const int r = r0 + wr * 16 + e + hi * 8;
      if (OUT_HEADS) {
        const int b = r >> 11, s = r & (SEQ - 1);
        const int h = c >> 6, d = c & 63;
        ((h16*)outp)[(((size_t)b * HEADS + h) * SEQ + s) * DEPTH + d] = (h16)acc[ct].f[e];
      } else {
        ((float*)outp)[(size_t)r * DIMW + c] = acc[ct].f[e];
      }
    }
  }
}

// ---------------------------------------------------------------------------
// Flash attention per (b, h, 128-row q tile). 256 threads = 8 waves; each wave
// owns 16 q rows so softmax row statistics stay within one half-wave.
// ---------------------------------------------------------------------------
__global__ __launch_bounds__(256) void flash_attn_kernel(
    const h16* __restrict__ qh, const h16* __restrict__ kh,
    const h16* __restrict__ vh, const float* __restrict__ mask,
    h16* __restrict__ att) {
  constexpr int XS = 72;
  __shared__ h16 lds_k[64 * XS];        // K tile row-major [j][d] (== B^T for QK^T)
  __shared__ h16 lds_vt[64 * XS];       // V tile transposed [d][j]
  __shared__ h16 lds_p[8 * 16 * XS];    // per-wave P staging [16][64]

  const int q0   = blockIdx.x * 128;
  const int h    = blockIdx.y;
  const int b    = blockIdx.z;
  const int tid  = threadIdx.x;
  const int lane = tid & 31;
  const int wv   = tid >> 5;
  const int nn   = lane & 15;
  const int hi   = lane >> 4;
  const size_t bh = (size_t)b * HEADS + h;

  // Preload Q A-fragments (reused every kv step): two b128 chunks per frag.
  const h16* qrow = qh + (bh * SEQ + q0 + wv * 16 + nn) * DEPTH;
  v16h qa0 = load_afrag(qrow, hi);
  v16h qa1 = load_afrag(qrow + 32, hi);

  float m_run[8], l_run[8];
#pragma unroll
  for (int e = 0; e < 8; ++e) { m_run[e] = -1e30f; l_run[e] = 0.0f; }
  Acc8 o_acc[4];
#pragma unroll
  for (int t = 0; t < 4; ++t) o_acc[t].v = (v8f){};

  for (int j0 = 0; j0 < SEQ; j0 += 64) {
    // ---- stage K via async global->LDS; V transposed via register scatter ----
    {
      const int jj = tid >> 2, seg = tid & 3;
      const h16* ks = kh + (bh * SEQ + j0 + jj) * DEPTH + seg * 16;
      async_copy_b128x2(ks, &lds_k[jj * XS + seg * 16]);

      const uint4* vs = (const uint4*)(vh + (bh * SEQ + j0 + jj) * DEPTH + seg * 16);
      V16 vu;
      vu.q[0] = vs[0];
      vu.q[1] = vs[1];
#pragma unroll
      for (int i = 0; i < 16; ++i)
        lds_vt[(seg * 16 + i) * XS + jj] = vu.hh[i];

      if (j0 + 64 < SEQ) {
        __builtin_prefetch(ks + 64 * DEPTH, 0, 0);
        __builtin_prefetch((const h16*)vs + 64 * DEPTH, 0, 0);
      }
    }
    async_wait0();
    __syncthreads();

    // ---- scores S = Q K^T over 4 kv column tiles ----
    Acc8 sc[4];
#pragma unroll
    for (int ct = 0; ct < 4; ++ct) {
      const h16* krow = &lds_k[(ct * 16 + nn) * XS];
      v16h b0 = load_bfrag(krow, hi);
      v16h b1 = load_bfrag(krow + 32, hi);
      v8f z = (v8f){};
      z = wmma_f16(qa0, b0, z);
      sc[ct].v = wmma_f16(qa1, b1, z);
    }

    // ---- scale + key-padding mask ----
    float mv[4];
#pragma unroll
    for (int ct = 0; ct < 4; ++ct)
      mv[ct] = mask[(size_t)b * SEQ + j0 + ct * 16 + nn] * 1e9f;
#pragma unroll
    for (int ct = 0; ct < 4; ++ct)
#pragma unroll
      for (int e = 0; e < 8; ++e)
        sc[ct].f[e] = sc[ct].f[e] * 0.125f - mv[ct];

    // ---- online softmax: row stats per accumulator slot + half-wave ----
    float mnew[8], alpha[8];
#pragma unroll
    for (int e = 0; e < 8; ++e) {
      float t = fmaxf(fmaxf(sc[0].f[e], sc[1].f[e]), fmaxf(sc[2].f[e], sc[3].f[e]));
#pragma unroll
      for (int mk = 1; mk < 16; mk <<= 1) t = fmaxf(t, __shfl_xor(t, mk, 32));
      mnew[e] = fmaxf(m_run[e], t);
      alpha[e] = __expf(m_run[e] - mnew[e]);
    }
#pragma unroll
    for (int e = 0; e < 8; ++e) {
      float s = 0.0f;
#pragma unroll
      for (int ct = 0; ct < 4; ++ct) {
        float p = __expf(sc[ct].f[e] - mnew[e]);
        sc[ct].f[e] = p;
        s += p;
      }
#pragma unroll
      for (int mk = 1; mk < 16; mk <<= 1) s += __shfl_xor(s, mk, 32);
      l_run[e] = l_run[e] * alpha[e] + s;
      m_run[e] = mnew[e];
    }
#pragma unroll
    for (int dt = 0; dt < 4; ++dt)
#pragma unroll
      for (int e = 0; e < 8; ++e) o_acc[dt].f[e] *= alpha[e];

    // ---- C-layout P -> A-layout via per-wave LDS region ----
    h16* pbase = &lds_p[wv * 16 * XS];
#pragma unroll
    for (int ct = 0; ct < 4; ++ct)
#pragma unroll
      for (int e = 0; e < 8; ++e)
        pbase[(e + hi * 8) * XS + ct * 16 + nn] = (h16)sc[ct].f[e];

    const h16* prow = pbase + nn * XS;
    v16h pa0 = load_afrag(prow, hi);
    v16h pa1 = load_afrag(prow + 32, hi);

    // ---- O += P V ----
#pragma unroll
    for (int dt = 0; dt < 4; ++dt) {
      const h16* vrow = &lds_vt[(dt * 16 + nn) * XS];
      v16h b0 = load_bfrag(vrow, hi);
      v16h b1 = load_bfrag(vrow + 32, hi);
      o_acc[dt].v = wmma_f16(pa0, b0, o_acc[dt].v);
      o_acc[dt].v = wmma_f16(pa1, b1, o_acc[dt].v);
    }
    __syncthreads();
  }

  // ---- normalize and store att (f16, [B,S,DIM] row-major) ----
#pragma unroll
  for (int dt = 0; dt < 4; ++dt) {
    const int d = dt * 16 + nn;
#pragma unroll
    for (int e = 0; e < 8; ++e) {
      const int s = q0 + wv * 16 + e + hi * 8;
      att[((size_t)b * SEQ + s) * DIMW + h * DEPTH + d] =
          (h16)(o_acc[dt].f[e] / l_run[e]);
    }
  }
}

extern "C" void kernel_launch(void* const* d_in, const int* in_sizes, int n_in,
                              void* d_out, int out_size, void* d_ws, size_t ws_size,
                              hipStream_t stream) {
  const float* q    = (const float*)d_in[0];
  const float* k    = (const float*)d_in[1];
  const float* v    = (const float*)d_in[2];
  const float* mask = (const float*)d_in[3];
  const float* WQ   = (const float*)d_in[4];
  const float* WK   = (const float*)d_in[5];
  const float* WV   = (const float*)d_in[6];
  const float* WO   = (const float*)d_in[7];
  float* out = (float*)d_out;

  const size_t elems = (size_t)BATCH * SEQ * DIMW;  // 8,388,608 halves each
  h16* qh  = (h16*)d_ws;
  h16* kh  = qh + elems;
  h16* vh  = kh + elems;
  h16* att = vh + elems;

  dim3 gblk(256);
  dim3 ggrid(DIMW / 128, (BATCH * SEQ) / 64);  // (8, 128)

  gemm_wmma_kernel<true,  true ><<<ggrid, gblk, 0, stream>>>(q, WQ, qh);
  gemm_wmma_kernel<true,  true ><<<ggrid, gblk, 0, stream>>>(k, WK, kh);
  gemm_wmma_kernel<true,  true ><<<ggrid, gblk, 0, stream>>>(v, WV, vh);

  dim3 ablk(256);
  dim3 agrid(SEQ / 128, HEADS, BATCH);         // (16, 16, 4)
  flash_attn_kernel<<<agrid, ablk, 0, stream>>>(qh, kh, vh, mask, att);

  gemm_wmma_kernel<false, false><<<ggrid, gblk, 0, stream>>>(att, WO, out);
}
